// DistributedMoELayer_59253368816243
// MI455X (gfx1250) — matrix operand order repaired and checked by reading, not compiled, not run on hardware
//
#include <hip/hip_runtime.h>
#include <hip/hip_bf16.h>

#define HIDDEN       1024
#define FFN_DIM      4096
#define NUM_EXPERTS  8
#define NUM_TOKENS   2048
#define M_TILE       32    // tokens per block (2 WMMA row-tiles)
#define FCHUNK       128
#define LDH          136   // padded bf16 row stride for the H1 LDS tile

typedef __attribute__((ext_vector_type(16))) __bf16 v16bf;
typedef __attribute__((ext_vector_type(8)))  float  v8f;

union Frag {
    v16bf          v;
    unsigned int   d[8];
};

__device__ __forceinline__ float gelu_tanh(float v) {
    return 0.5f * v * (1.0f + tanhf(0.7978845608028654f * (v + 0.044715f * v * v * v)));
}

// ---------------------------------------------------------------------------
// Kernel 0: zero the output and the per-expert counters (runs every call).
// ---------------------------------------------------------------------------
__global__ void moe_zero_kernel(float* __restrict__ out, int n_out, int* __restrict__ cnt) {
    int i = blockIdx.x * blockDim.x + threadIdx.x;
    if (i < n_out) out[i] = 0.0f;
    if (i < NUM_EXPERTS) cnt[i] = 0;
}

// ---------------------------------------------------------------------------
// Kernel 1: router. One wave32 per token: logits = x @ router_w, softmax,
// top-2, renormalize; scatter (token, weight) into per-expert lists.
// ---------------------------------------------------------------------------
__global__ __launch_bounds__(256) void moe_router_kernel(
    const float* __restrict__ x, const float* __restrict__ rw,
    int* __restrict__ cnt, int* __restrict__ idxl, float* __restrict__ wgtl)
{
    int token = (blockIdx.x * blockDim.x + threadIdx.x) >> 5;
    int lane  = threadIdx.x & 31;
    if (token >= NUM_TOKENS) return;

    const float* xr = x + (size_t)token * HIDDEN;
    float acc[NUM_EXPERTS];
#pragma unroll
    for (int e = 0; e < NUM_EXPERTS; ++e) acc[e] = 0.0f;

    for (int h = lane; h < HIDDEN; h += 32) {
        float xv = xr[h];
        const float* r = rw + (size_t)h * NUM_EXPERTS;
#pragma unroll
        for (int e = 0; e < NUM_EXPERTS; ++e) acc[e] = fmaf(xv, r[e], acc[e]);
    }
#pragma unroll
    for (int e = 0; e < NUM_EXPERTS; ++e) {
#pragma unroll
        for (int s = 16; s > 0; s >>= 1) acc[e] += __shfl_xor(acc[e], s, 32);
    }

    if (lane == 0) {
        int i1 = 0;
#pragma unroll
        for (int e = 1; e < NUM_EXPERTS; ++e) if (acc[e] > acc[i1]) i1 = e;
        int i2 = (i1 == 0) ? 1 : 0;
#pragma unroll
        for (int e = 0; e < NUM_EXPERTS; ++e)
            if (e != i1 && acc[e] > acc[i2]) i2 = e;

        // softmax over top-2 logits == full softmax then renormalize top-2
        float l1 = acc[i1], l2 = acc[i2];
        float m  = fmaxf(l1, l2);
        float e1 = __expf(l1 - m), e2 = __expf(l2 - m);
        float inv = 1.0f / (e1 + e2);

        int s1 = atomicAdd(&cnt[i1], 1);
        idxl[i1 * NUM_TOKENS + s1] = token;
        wgtl[i1 * NUM_TOKENS + s1] = e1 * inv;
        int s2 = atomicAdd(&cnt[i2], 1);
        idxl[i2 * NUM_TOKENS + s2] = token;
        wgtl[i2 * NUM_TOKENS + s2] = e2 * inv;
    }
}

// ---------------------------------------------------------------------------
// Kernel 2: fused expert FFN. block = 8 waves, tile = 32 gathered tokens.
// Each B-fragment (weights) is shared by two WMMAs (rows 0-15 and 16-31),
// halving HBM weight traffic vs a 16-row tile.
// ---------------------------------------------------------------------------
__global__ __launch_bounds__(256) void moe_ffn_kernel(
    const float* __restrict__ x,  const float* __restrict__ w1,
    const float* __restrict__ w2, const int*   __restrict__ cnt,
    const int*   __restrict__ idxl, const float* __restrict__ wgtl,
    float* __restrict__ out)
{
    const int e    = blockIdx.y;
    const int tile = blockIdx.x;
    const int n_e  = cnt[e];
    if (tile * M_TILE >= n_e) return;

    __shared__ __align__(16) __bf16 Xl[M_TILE * HIDDEN]; // 64 KB bf16 X tile
    __shared__ __align__(16) __bf16 Hl[M_TILE * LDH];    // bf16 H1 chunk tile
    __shared__ int   tokS[M_TILE];
    __shared__ float wS[M_TILE];

    const int tid  = threadIdx.x;
    const int wv   = tid >> 5;
    const int lane = tid & 31;
    const int ml   = lane & 15;                  // matrix row / column sub-index
    const int koff = (lane < 16) ? 0 : 8;        // A-frag K sub-offset (ISA layout)
    const int kb16 = (lane < 16) ? 0 : 16;       // unused; B-frags use koff (same k order as A)

    if (tid < M_TILE) {
        int r  = tile * M_TILE + tid;
        bool ok = r < n_e;
        tokS[tid] = ok ? idxl[e * NUM_TOKENS + r] : -1;
        wS[tid]   = ok ? wgtl[e * NUM_TOKENS + r] : 0.0f;
    }
    __syncthreads();

    // gather X rows -> bf16 LDS (invalid rows zero-filled)
    for (int i = tid; i < M_TILE * HIDDEN; i += 256) {
        int r = i >> 10, c = i & (HIDDEN - 1);
        int t = tokS[r];
        float v = (t >= 0) ? x[(size_t)t * HIDDEN + c] : 0.0f;
        Xl[i] = (__bf16)v;
    }
    __syncthreads();

    const float* W1e = w1 + (size_t)e * HIDDEN * FFN_DIM;
    const float* W2e = w2 + (size_t)e * FFN_DIM * HIDDEN;

    v8f accLo[8], accHi[8];
#pragma unroll
    for (int j = 0; j < 8; ++j) { accLo[j] = (v8f){}; accHi[j] = (v8f){}; }

    const int nwave = wv * 128;   // this wave's 128 output columns

    for (int fc = 0; fc < FFN_DIM; fc += FCHUNK) {
        // ---- phase 1: two 16x16 tiles of gelu(X @ W1) per wave ----
        v8f c1lo = (v8f){}, c1hi = (v8f){};
        const int ncol1 = fc + wv * 16 + ml;
        for (int k = 0; k < HIDDEN; k += 32) {
            Frag aLo, aHi;
            {
                const unsigned int* p0 = (const unsigned int*)
                    __builtin_assume_aligned(&Xl[ml * HIDDEN + k + koff], 16);
                const unsigned int* p1 = (const unsigned int*)
                    __builtin_assume_aligned(&Xl[ml * HIDDEN + k + koff + 16], 16);
                const unsigned int* p2 = (const unsigned int*)
                    __builtin_assume_aligned(&Xl[(16 + ml) * HIDDEN + k + koff], 16);
                const unsigned int* p3 = (const unsigned int*)
                    __builtin_assume_aligned(&Xl[(16 + ml) * HIDDEN + k + koff + 16], 16);
#pragma unroll
                for (int q = 0; q < 4; ++q) {
                    aLo.d[q] = p0[q]; aLo.d[4 + q] = p1[q];
                    aHi.d[q] = p2[q]; aHi.d[4 + q] = p3[q];
                }
            }
            Frag b;
            {
                const float* bp = W1e + (size_t)(k + koff) * FFN_DIM + ncol1;
                __builtin_prefetch(bp + (size_t)32 * FFN_DIM, 0, 1);
#pragma unroll
                for (int kk = 0; kk < 16; ++kk)
                    b.v[kk] = (__bf16)bp[(size_t)(kk + (kk < 8 ? 0 : 8)) * FFN_DIM];   // hw cvt, RNE
            }
            c1lo = __builtin_amdgcn_wmma_f32_16x16x32_bf16(
                       false, aLo.v, false, b.v, (short)0, c1lo, false, false);
            c1hi = __builtin_amdgcn_wmma_f32_16x16x32_bf16(
                       false, aHi.v, false, b.v, (short)0, c1hi, false, false);
        }
#pragma unroll
        for (int j = 0; j < 8; ++j) {
            int row = (lane < 16) ? j : (j + 8);
            int col = wv * 16 + ml;
            Hl[row * LDH + col]        = (__bf16)gelu_tanh(c1lo[j]);
            Hl[(16 + row) * LDH + col] = (__bf16)gelu_tanh(c1hi[j]);
        }
        __syncthreads();

        // ---- phase 2: acc += H1_chunk(32x128) @ W2[chunk, wave's 128 cols] ----
#pragma unroll
        for (int jt = 0; jt < 8; ++jt) {
            const int nc = nwave + jt * 16 + ml;
#pragma unroll
            for (int k2 = 0; k2 < FCHUNK; k2 += 32) {
                Frag a2lo, a2hi;
                {
                    const unsigned int* p0 = (const unsigned int*)
                        __builtin_assume_aligned(&Hl[ml * LDH + k2 + koff], 16);
                    const unsigned int* p1 = (const unsigned int*)
                        __builtin_assume_aligned(&Hl[ml * LDH + k2 + koff + 16], 16);
                    const unsigned int* p2 = (const unsigned int*)
                        __builtin_assume_aligned(&Hl[(16 + ml) * LDH + k2 + koff], 16);
                    const unsigned int* p3 = (const unsigned int*)
                        __builtin_assume_aligned(&Hl[(16 + ml) * LDH + k2 + koff + 16], 16);
#pragma unroll
                    for (int q = 0; q < 4; ++q) {
                        a2lo.d[q] = p0[q]; a2lo.d[4 + q] = p1[q];
                        a2hi.d[q] = p2[q]; a2hi.d[4 + q] = p3[q];
                    }
                }
                Frag b2;
                {
                    const float* bp = W2e + (size_t)(fc + k2 + koff) * HIDDEN + nc;
#pragma unroll
                    for (int kk = 0; kk < 16; ++kk)
                        b2.v[kk] = (__bf16)bp[(size_t)(kk + (kk < 8 ? 0 : 8)) * HIDDEN];
                }
                accLo[jt] = __builtin_amdgcn_wmma_f32_16x16x32_bf16(
                                false, a2lo.v, false, b2.v, (short)0, accLo[jt], false, false);
                accHi[jt] = __builtin_amdgcn_wmma_f32_16x16x32_bf16(
                                false, a2hi.v, false, b2.v, (short)0, accHi[jt], false, false);
            }
        }
        __syncthreads();
    }

    // ---- epilogue: weighted scatter-add into the shared output ----
#pragma unroll
    for (int jt = 0; jt < 8; ++jt) {
#pragma unroll
        for (int i = 0; i < 8; ++i) {
            int row = (lane < 16) ? i : (i + 8);
            int col = nwave + jt * 16 + ml;
            int tL  = tokS[row];
            if (tL >= 0)
                atomicAdd(&out[(size_t)tL * HIDDEN + col], wS[row] * accLo[jt][i]);
            int tH  = tokS[16 + row];
            if (tH >= 0)
                atomicAdd(&out[(size_t)tH * HIDDEN + col], wS[16 + row] * accHi[jt][i]);
        }
    }
}

// ---------------------------------------------------------------------------
extern "C" void kernel_launch(void* const* d_in, const int* in_sizes, int n_in,
                              void* d_out, int out_size, void* d_ws, size_t ws_size,
                              hipStream_t stream) {
    const float* x  = (const float*)d_in[0];
    const float* rw = (const float*)d_in[1];
    const float* w1 = (const float*)d_in[2];
    const float* w2 = (const float*)d_in[3];
    float* out = (float*)d_out;

    // workspace layout: [cnt: 8 ints][idx: E*T ints][wgt: E*T floats] (~131 KB)
    int*   cnt  = (int*)d_ws;
    int*   idxl = cnt + 8;
    float* wgtl = (float*)(idxl + NUM_EXPERTS * NUM_TOKENS);

    moe_zero_kernel<<<(out_size + 255) / 256, 256, 0, stream>>>(out, out_size, cnt);

    moe_router_kernel<<<NUM_TOKENS / 8, 256, 0, stream>>>(x, rw, cnt, idxl, wgtl);

    dim3 grid(NUM_TOKENS / M_TILE, NUM_EXPERTS);
    moe_ffn_kernel<<<grid, 256, 0, stream>>>(x, w1, w2, cnt, idxl, wgtl, out);
}
